// PrometheusBlockFT_80711025426906
// MI455X (gfx1250) — compile-verified, matrix operand order checked by reading, not run on hardware
//
#include <hip/hip_runtime.h>
#include <hip/hip_bf16.h>
#include <math.h>

typedef __bf16 bf16;
typedef __bf16 v16bf __attribute__((ext_vector_type(16)));
typedef float  v8f   __attribute__((ext_vector_type(8)));

union FragU { float4 f4[2]; v16bf v; };

__device__ __forceinline__ v16bf frag_ld(const bf16* p) {
  FragU u;
  u.f4[0] = *(const float4*)(p);
  u.f4[1] = *(const float4*)(p + 16);
  return u.v;
}

// Async copy of 32 bytes: global [gp, gp+32) -> LDS [lds, lds+32).
// INST_OFFSET is applied to BOTH the LDS and global addresses (ISA 15.18.3).
__device__ __forceinline__ void async_cp32(unsigned lds, const bf16* gp) {
  asm volatile(
      "global_load_async_to_lds_b128 %0, %1, off\n\t"
      "global_load_async_to_lds_b128 %0, %1, off offset:16"
      :: "v"(lds), "v"(gp) : "memory");
}

__device__ __forceinline__ void wait_async0() {
  asm volatile("s_wait_asynccnt 0x0" ::: "memory");
}

__device__ __forceinline__ unsigned lds_off(const void* p) {
  // __shared__ generic pointer: low 32 bits are the wave-relative LDS address.
  return (unsigned)(size_t)p;
}

// ---------------------------------------------------------------------------
// Universal bf16 WMMA GEMM:  C = epilogue( scale * (A[M,K] @ Bt[N,K]^T) )
// BM=128, BN=128, BK=32, 256 threads = 8 wave32, each wave 32x64 (2x4 WMMA).
// Tiles are staged with GLOBAL_LOAD_ASYNC_TO_LDS_B128 (ASYNCcnt), single
// barrier per K-step.
// ---------------------------------------------------------------------------
template<bool GELU, bool BIASF, bool ACCUM, bool ROWW, bool GATHER, bool EXPERT,
         bool WF32, bool WBF16>
__global__ __launch_bounds__(256)
void gemm_wmma(const bf16* __restrict__ A, const bf16* __restrict__ Bt,
               float* __restrict__ C, bf16* __restrict__ Cb,
               const float* __restrict__ bias, const float* __restrict__ roww,
               const int* __restrict__ perm, const int* __restrict__ tileExp,
               const bf16* __restrict__ zrow,
               float scale, int M, int N, int K, long expStrideB, int biasN)
{
  (void)M;
  __shared__ bf16 sA[2][128 * 40];
  __shared__ bf16 sB[2][128 * 40];

  const int tid = threadIdx.x;
  const int m0 = blockIdx.y * 128;
  const int n0 = blockIdx.x * 128;

  int e = 0;
  if (EXPERT) e = tileExp[blockIdx.y];
  const bf16* Bte = Bt + (EXPERT ? (long)e * expStrideB : 0);
  const float* biasE = BIASF ? (bias + (EXPERT ? (long)e * biasN : 0)) : nullptr;

  // global -> LDS loader: thread t covers row t>>1, 16 cols starting (t&1)*16
  const int lr = tid >> 1;
  const int lc = (tid & 1) * 16;
  bool aValid = true;
  long arow;
  if (GATHER) {
    int s = perm[m0 + lr];
    aValid = (s >= 0);
    arow = aValid ? (long)s : 0;
  } else {
    arow = m0 + lr;
  }
  const bf16* aPtr = A + arow * (long)K + lc;
  const bf16* bPtr = Bte + (long)(n0 + lr) * K + lc;

  const unsigned la0 = lds_off(&sA[0][lr * 40 + lc]);
  const unsigned la1 = lds_off(&sA[1][lr * 40 + lc]);
  const unsigned lb0 = lds_off(&sB[0][lr * 40 + lc]);
  const unsigned lb1 = lds_off(&sB[1][lr * 40 + lc]);

  auto issue = [&](int kt) {
    const int buf = kt & 1;
    const bf16* ap = (!GATHER || aValid) ? (aPtr + kt * 32) : zrow;
    const bf16* bp = bPtr + kt * 32;
    async_cp32(buf ? la1 : la0, ap);
    async_cp32(buf ? lb1 : lb0, bp);
  };

  const int lane = tid & 31;
  const int wv = tid >> 5;
  const int wm = (wv >> 1) * 32;   // wave M offset (4 waves in M)
  const int wn = (wv & 1) * 64;    // wave N offset (2 waves in N)
  const int fr = lane & 15;
  const int kb = (lane >> 4) * 8;  // ISA 16-bit A/B fragment k base

  v8f acc[2][4] = {};

  auto compute = [&](int buf) {
    v16bf af[2], bfr[4];
#pragma unroll
    for (int i = 0; i < 2; ++i)
      af[i] = frag_ld(&sA[buf][(wm + i * 16 + fr) * 40 + kb]);
#pragma unroll
    for (int j = 0; j < 4; ++j)
      bfr[j] = frag_ld(&sB[buf][(wn + j * 16 + fr) * 40 + kb]);
#pragma unroll
    for (int i = 0; i < 2; ++i)
#pragma unroll
      for (int j = 0; j < 4; ++j)
        acc[i][j] = __builtin_amdgcn_wmma_f32_16x16x32_bf16(
            false, af[i], false, bfr[j], (short)0, acc[i][j], false, false);
  };

  const int nk = K >> 5;
  issue(0);
  for (int kt = 0; kt < nk; ++kt) {
    wait_async0();        // this wave's async stores to LDS complete
    __syncthreads();      // publish across all waves
    if (kt + 1 < nk) issue(kt + 1);  // other buffer; safe: last read pre-barrier
    compute(kt & 1);
  }

  // Epilogue. C/D layout: row = v + 8*(lane>>4), col = lane&15 within 16x16.
#pragma unroll
  for (int i = 0; i < 2; ++i) {
#pragma unroll
    for (int v = 0; v < 8; ++v) {
      const int gR = m0 + wm + i * 16 + v + ((lane >> 4) << 3);
      float rwv = 1.0f;
      if (ROWW) rwv = roww[gR];
#pragma unroll
      for (int j = 0; j < 4; ++j) {
        const int gC = n0 + wn + j * 16 + fr;
        float t = acc[i][j][v] * scale;
        if (BIASF) t += biasE[gC];
        if (GELU) t = 0.5f * t * (1.0f + erff(t * 0.70710678118654752f));
        if (ROWW) t *= rwv;
        const long idx = (long)gR * N + gC;
        if (ACCUM) C[idx] += t;
        else if (WF32) C[idx] = t;
        if (WBF16) Cb[idx] = (bf16)t;
      }
    }
  }
}

// ---------------------------------------------------------------------------
// LayerNorm: one block per token, writes f32 + bf16 mirror.
// ---------------------------------------------------------------------------
__global__ __launch_bounds__(256)
void ln_kernel(const float* __restrict__ x, const float* __restrict__ g,
               const float* __restrict__ b, float* __restrict__ xn,
               bf16* __restrict__ xnb)
{
  const int t = blockIdx.x;
  const float* row = x + (long)t * 1024;
  float s = 0.f, ss = 0.f;
  for (int i = threadIdx.x; i < 1024; i += 256) {
    float v = row[i];
    s += v;
    ss += v * v;
  }
  for (int m = 16; m; m >>= 1) {
    s += __shfl_xor(s, m, 32);
    ss += __shfl_xor(ss, m, 32);
  }
  __shared__ float rs[8], rss[8];
  const int wv = threadIdx.x >> 5;
  if ((threadIdx.x & 31) == 0) { rs[wv] = s; rss[wv] = ss; }
  __syncthreads();
  if (threadIdx.x == 0) {
    float S = 0.f, SS = 0.f;
    for (int i = 0; i < 8; ++i) { S += rs[i]; SS += rss[i]; }
    rs[0] = S; rss[0] = SS;
  }
  __syncthreads();
  const float mu = rs[0] * (1.0f / 1024.0f);
  const float var = rss[0] * (1.0f / 1024.0f) - mu * mu;
  const float inv = rsqrtf(var + 1e-5f);
  for (int i = threadIdx.x; i < 1024; i += 256) {
    float v = (row[i] - mu) * inv * g[i] + b[i];
    xn[(long)t * 1024 + i] = v;
    xnb[(long)t * 1024 + i] = (bf16)v;
  }
}

// ---------------------------------------------------------------------------
// Gate: one wave32 per token; logits over 6 experts, top-2 softmax, rank.
// ---------------------------------------------------------------------------
__global__ __launch_bounds__(256)
void gate_kernel(const float* __restrict__ xn, const float* __restrict__ gw,
                 const float* __restrict__ gb, int* __restrict__ eidx,
                 float* __restrict__ ewgt, int* __restrict__ rnk,
                 int* __restrict__ cnt)
{
  const int lane = threadIdx.x & 31;
  const int wv = threadIdx.x >> 5;
  const int t = blockIdx.x * 8 + wv;
  const float* row = xn + (long)t * 1024;
  float l[6] = {0.f, 0.f, 0.f, 0.f, 0.f, 0.f};
  for (int d = lane; d < 1024; d += 32) {
    const float v = row[d];
    const float* wp = gw + d * 6;
#pragma unroll
    for (int e = 0; e < 6; ++e) l[e] += v * wp[e];
  }
#pragma unroll
  for (int e = 0; e < 6; ++e)
    for (int m = 16; m; m >>= 1) l[e] += __shfl_xor(l[e], m, 32);
  if (lane == 0) {
#pragma unroll
    for (int e = 0; e < 6; ++e) l[e] += gb[e];
    int e0 = 0;
    for (int e = 1; e < 6; ++e) if (l[e] > l[e0]) e0 = e;
    int e1 = (e0 == 0) ? 1 : 0;
    for (int e = 0; e < 6; ++e) if (e != e0 && l[e] > l[e1]) e1 = e;
    float w1 = __expf(l[e1] - l[e0]);
    const float s = 1.0f + w1;
    const float w0 = 1.0f / s;
    w1 = w1 / s;
    const int r0 = atomicAdd(&cnt[e0], 1);
    const int r1 = atomicAdd(&cnt[e1], 1);
    eidx[2 * t] = e0; eidx[2 * t + 1] = e1;
    ewgt[2 * t] = w0; ewgt[2 * t + 1] = w1;
    rnk[2 * t] = r0;  rnk[2 * t + 1] = r1;
  }
}

__global__ void init_kernel(int* perm, float* rw, int* cnt, int* tileE, int* zrow)
{
  const int i = blockIdx.x * 256 + threadIdx.x;
  if (i < 17152) { perm[i] = -1; rw[i] = 0.0f; }
  if (i < 6) cnt[i] = 0;
  if (i < 134) tileE[i] = 0;
  if (i < 16) zrow[i] = 0;   // 64B zero page for gather padding rows
}

__global__ void offsets_kernel(const int* __restrict__ cnt, int* __restrict__ off,
                               int* __restrict__ tileE)
{
  if (threadIdx.x == 0) {
    int o = 0;
    for (int e = 0; e < 6; ++e) {
      off[e] = o;
      const int pad = (cnt[e] + 127) & ~127;
      const int t0 = o >> 7, t1 = (o + pad) >> 7;
      for (int tt = t0; tt < t1; ++tt) tileE[tt] = e;
      o += pad;
    }
    off[6] = o;
  }
}

__global__ void scatter_kernel(const int* __restrict__ eidx,
                               const float* __restrict__ ewgt,
                               const int* __restrict__ rnk,
                               const int* __restrict__ off,
                               int* __restrict__ perm, float* __restrict__ rw,
                               int* __restrict__ slot)
{
  const int i = blockIdx.x * 256 + threadIdx.x;
  if (i >= 16384) return;
  const int e = eidx[i];
  const int pos = off[e] + rnk[i];
  perm[pos] = i >> 1;
  rw[pos] = ewgt[i];
  slot[i] = pos;
}

__global__ __launch_bounds__(256)
void combine_kernel(const float* __restrict__ eo, const int* __restrict__ slot,
                    bf16* __restrict__ moeb)
{
  const int t = blockIdx.x;
  const long p0 = (long)slot[2 * t] * 1024;
  const long p1 = (long)slot[2 * t + 1] * 1024;
  for (int d = threadIdx.x; d < 1024; d += 256)
    moeb[(long)t * 1024 + d] = (bf16)(eo[p0 + d] + eo[p1 + d]);
}

__global__ __launch_bounds__(256)
void final_kernel(const float* __restrict__ x, const float* __restrict__ abel,
                  const float* __restrict__ nonab, const float* __restrict__ gcw,
                  const float* __restrict__ gcb, float* __restrict__ out)
{
  const int t = blockIdx.x;
  const long base = (long)t * 1024;
  float s = 0.f;
  for (int d = threadIdx.x; d < 1024; d += 256)
    s += abel[base + d] * gcw[d] + nonab[base + d] * gcw[1024 + d];
  for (int m = 16; m; m >>= 1) s += __shfl_xor(s, m, 32);
  __shared__ float r[8];
  if ((threadIdx.x & 31) == 0) r[threadIdx.x >> 5] = s;
  __syncthreads();
  if (threadIdx.x == 0) {
    float S = 0.f;
    for (int i = 0; i < 8; ++i) S += r[i];
    r[0] = 1.0f / (1.0f + __expf(-(S + gcb[0])));
  }
  __syncthreads();
  const float gmix = r[0];
  for (int d = threadIdx.x; d < 1024; d += 256) {
    const float a = abel[base + d];
    const float nb = nonab[base + d];
    out[base + d] = x[base + d] + a * (1.0f - gmix) + nb * gmix;
  }
}

__global__ void cvt_rm_kernel(const float* __restrict__ in, bf16* __restrict__ out,
                              float scale, long n)
{
  const long i = (long)blockIdx.x * 256 + threadIdx.x;
  if (i < n) out[i] = (bf16)(in[i] * scale);
}

// out[b][c][r] = in[b][r][c] * scale  (produces Bt = B^T in [N][K] layout)
__global__ void cvt_tr_kernel(const float* __restrict__ in, bf16* __restrict__ out,
                              float scale, int R, int C)
{
  const long base = (long)blockIdx.y * R * C;
  const int i = blockIdx.x * 256 + threadIdx.x;
  if (i < R * C) {
    const int r = i / C, c = i % C;
    out[base + (long)c * R + r] = (bf16)(in[base + i] * scale);
  }
}

__global__ void expm_init_kernel(const float* __restrict__ G, float scale,
                                 float* __restrict__ acc)
{
  const int i = blockIdx.x * 256 + threadIdx.x;
  const int r = i >> 10, c = i & 1023;
  acc[i] = G[i] * scale + ((r == c) ? 1.0f : 0.0f);
}

// ---------------------------------------------------------------------------
extern "C" void kernel_launch(void* const* d_in, const int* in_sizes, int n_in,
                              void* d_out, int out_size, void* d_ws, size_t ws_size,
                              hipStream_t stream)
{
  (void)in_sizes; (void)n_in; (void)out_size; (void)ws_size;
  const float* x      = (const float*)d_in[0];
  const float* ln_g   = (const float*)d_in[1];
  const float* ln_b   = (const float*)d_in[2];
  const float* aw1    = (const float*)d_in[3];
  const float* ab1    = (const float*)d_in[4];
  const float* aw2    = (const float*)d_in[5];
  const float* ab2    = (const float*)d_in[6];
  const float* gate_w = (const float*)d_in[7];
  const float* gate_b = (const float*)d_in[8];
  const float* ew1    = (const float*)d_in[9];
  const float* eb1    = (const float*)d_in[10];
  const float* ew2    = (const float*)d_in[11];
  const float* eb2    = (const float*)d_in[12];
  const float* gens   = (const float*)d_in[13];
  const float* gc_w   = (const float*)d_in[14];
  const float* gc_b   = (const float*)d_in[15];
  float* out = (float*)d_out;

  constexpr int T = 8192, D = 1024, A4 = 4096, H = 2048, E = 6;
  constexpr int NPAD = 17152, NTILES = 134;   // 134*128 rows, top-2 * 8192 + padding

  char* base = (char*)d_ws;
  size_t off = 0;
  auto alloc = [&](size_t bytes) -> void* {
    void* p = base + off;
    off += (bytes + 255) & ~(size_t)255;
    return p;
  };
  auto cdiv = [](long a, long b) { return (int)((a + b - 1) / b); };

  float* xn    = (float*)alloc((size_t)T * D * 4);
  bf16*  xnb   = (bf16*) alloc((size_t)T * D * 2);
  bf16*  aw1t  = (bf16*) alloc((size_t)D * A4 * 2);
  bf16*  aw2t  = (bf16*) alloc((size_t)A4 * D * 2);
  bf16*  ew1t  = (bf16*) alloc((size_t)E * D * H * 2);
  bf16*  ew2t  = (bf16*) alloc((size_t)E * H * D * 2);
  bf16*  hbig  = (bf16*) alloc((size_t)NPAD * H * 2);   // reused: h1 (abel) then h_moe
  float* abel  = (float*)alloc((size_t)T * D * 4);
  float* eo    = (float*)alloc((size_t)NPAD * D * 4);
  bf16*  moeb  = (bf16*) alloc((size_t)T * D * 2);
  float* nonab = (float*)alloc((size_t)T * D * 4);
  int*   cnt   = (int*)  alloc(E * 4);
  int*   offs  = (int*)  alloc(8 * 4);
  int*   tileE = (int*)  alloc(NTILES * 4);
  int*   eidx  = (int*)  alloc(2 * T * 4);
  float* ewgt  = (float*)alloc(2 * T * 4);
  int*   rnk   = (int*)  alloc(2 * T * 4);
  int*   perm  = (int*)  alloc(NPAD * 4);
  float* rw    = (float*)alloc(NPAD * 4);
  int*   slot  = (int*)  alloc(2 * T * 4);
  int*   zrow  = (int*)  alloc(256);               // 64B zero page (+ padding)
  float* accA  = (float*)alloc((size_t)D * D * 4);
  float* accB  = (float*)alloc((size_t)D * D * 4);
  float* Mcur  = (float*)alloc((size_t)D * D * 4);
  float* Mnext = (float*)alloc((size_t)D * D * 4);
  bf16*  Gpt   = (bf16*) alloc((size_t)D * D * 2);
  bf16*  tA    = (bf16*) alloc((size_t)D * D * 2);
  bf16*  tB    = (bf16*) alloc((size_t)D * D * 2);
  bf16*  Prm   = (bf16*) alloc((size_t)D * D * 2);
  bf16*  Ptr   = (bf16*) alloc((size_t)D * D * 2);
  bf16*  Mt    = (bf16*) alloc((size_t)D * D * 2);

  // 1) LayerNorm (f32 + bf16 mirror)
  ln_kernel<<<T, 256, 0, stream>>>(x, ln_g, ln_b, xn, xnb);

  // 2) Weight conversions to bf16 [N][K] transposed layout
  cvt_tr_kernel<<<dim3(cdiv((long)D * A4, 256), 1), 256, 0, stream>>>(aw1, aw1t, 1.0f, D, A4);
  cvt_tr_kernel<<<dim3(cdiv((long)A4 * D, 256), 1), 256, 0, stream>>>(aw2, aw2t, 1.0f, A4, D);
  cvt_tr_kernel<<<dim3(cdiv((long)D * H, 256), E), 256, 0, stream>>>(ew1, ew1t, 1.0f, D, H);
  cvt_tr_kernel<<<dim3(cdiv((long)H * D, 256), E), 256, 0, stream>>>(ew2, ew2t, 1.0f, H, D);

  // 3) Abel branch: h1 = gelu(xn@aw1+ab1) [bf16], abel = h1@aw2+ab2 [f32]
  gemm_wmma<true, true, false, false, false, false, false, true>
      <<<dim3(A4 / 128, T / 128), 256, 0, stream>>>(
          xnb, aw1t, nullptr, hbig, ab1, nullptr, nullptr, nullptr, nullptr,
          1.0f, T, A4, D, 0, 0);
  gemm_wmma<false, true, false, false, false, false, true, false>
      <<<dim3(D / 128, T / 128), 256, 0, stream>>>(
          hbig, aw2t, abel, nullptr, ab2, nullptr, nullptr, nullptr, nullptr,
          1.0f, T, D, A4, 0, 0);

  // 4) Routing (top-2 of 6 experts, segment-padded to 128 rows)
  init_kernel<<<cdiv(NPAD, 256), 256, 0, stream>>>(perm, rw, cnt, tileE, zrow);
  gate_kernel<<<T / 8, 256, 0, stream>>>(xn, gate_w, gate_b, eidx, ewgt, rnk, cnt);
  offsets_kernel<<<1, 1, 0, stream>>>(cnt, offs, tileE);
  scatter_kernel<<<cdiv(2 * T, 256), 256, 0, stream>>>(eidx, ewgt, rnk, offs, perm, rw, slot);

  // 5) Sparse MoE: gathered GEMM1 (gelu, bf16) then weighted GEMM2 (f32)
  gemm_wmma<true, true, false, false, true, true, false, true>
      <<<dim3(H / 128, NTILES), 256, 0, stream>>>(
          xnb, ew1t, nullptr, hbig, eb1, nullptr, perm, tileE, (const bf16*)zrow,
          1.0f, NPAD, H, D, (long)H * D, H);
  gemm_wmma<false, true, false, true, false, true, true, false>
      <<<dim3(D / 128, NTILES), 256, 0, stream>>>(
          hbig, ew2t, eo, nullptr, eb2, rw, nullptr, tileE, nullptr,
          1.0f, NPAD, D, H, (long)D * H, D);
  combine_kernel<<<T, 256, 0, stream>>>(eo, slot, moeb);

  // 6) expm(G_i) via scaling (2^-3) + Taylor(6) + 3 squarings; fold into M
  const float S = 0.125f;
  const int g1 = cdiv((long)D * D, 256);
  for (int i = 0; i < 4; ++i) {
    const float* Gi = gens + (size_t)i * D * D;
    expm_init_kernel<<<g1, 256, 0, stream>>>(Gi, S, accA);
    cvt_rm_kernel<<<g1, 256, 0, stream>>>(Gi, tA, S, (long)D * D);
    cvt_tr_kernel<<<dim3(g1, 1), 256, 0, stream>>>(Gi, Gpt, S, D, D);
    bf16* ta = tA; bf16* tb = tB;
    for (int k = 2; k <= 6; ++k) {
      gemm_wmma<false, false, true, false, false, false, false, true>
          <<<dim3(D / 128, D / 128), 256, 0, stream>>>(
              ta, Gpt, accA, tb, nullptr, nullptr, nullptr, nullptr, nullptr,
              1.0f / (float)k, D, D, D, 0, 0);
      bf16* tmp = ta; ta = tb; tb = tmp;
    }
    float* cur = accA; float* oth = accB;
    for (int sq = 0; sq < 3; ++sq) {
      cvt_rm_kernel<<<g1, 256, 0, stream>>>(cur, Prm, 1.0f, (long)D * D);
      cvt_tr_kernel<<<dim3(g1, 1), 256, 0, stream>>>(cur, Ptr, 1.0f, D, D);
      gemm_wmma<false, false, false, false, false, false, true, false>
          <<<dim3(D / 128, D / 128), 256, 0, stream>>>(
              Prm, Ptr, oth, nullptr, nullptr, nullptr, nullptr, nullptr, nullptr,
              1.0f, D, D, D, 0, 0);
      float* t2 = cur; cur = oth; oth = t2;
    }
    if (i == 0) {
      hipMemcpyAsync(Mcur, cur, (size_t)D * D * 4, hipMemcpyDeviceToDevice, stream);
    } else {
      cvt_rm_kernel<<<g1, 256, 0, stream>>>(Mcur, Prm, 1.0f, (long)D * D);
      cvt_tr_kernel<<<dim3(g1, 1), 256, 0, stream>>>(cur, Ptr, 1.0f, D, D);
      gemm_wmma<false, false, false, false, false, false, true, false>
          <<<dim3(D / 128, D / 128), 256, 0, stream>>>(
              Prm, Ptr, Mnext, nullptr, nullptr, nullptr, nullptr, nullptr, nullptr,
              1.0f, D, D, D, 0, 0);
      float* t2 = Mcur; Mcur = Mnext; Mnext = t2;
    }
  }
  cvt_tr_kernel<<<dim3(g1, 1), 256, 0, stream>>>(Mcur, Mt, 1.0f, D, D);

  // 7) Rotation: nonabel = moe @ M  (single GEMM instead of 4)
  gemm_wmma<false, false, false, false, false, false, true, false>
      <<<dim3(D / 128, T / 128), 256, 0, stream>>>(
          moeb, Mt, nonab, nullptr, nullptr, nullptr, nullptr, nullptr, nullptr,
          1.0f, T, D, D, 0, 0);

  // 8) Sigmoid-gated mix + residual
  final_kernel<<<T, 256, 0, stream>>>(x, abel, nonab, gc_w, gc_b, out);
}
